// SimpleBiLSTM_47974784696737
// MI455X (gfx1250) — matrix operand order repaired and checked by reading, not compiled, not run on hardware
//
#include <hip/hip_runtime.h>
#include <hip/hip_bf16.h>

// ---------------------------------------------------------------------------
// BiLSTM (2-layer, bidirectional) for MI455X / gfx1250.
//   VOCAB=50000, E=H=256, B=64, T=512, NCLASS=5
// Strategy:
//   K0a: embedding gather -> bf16 xs[T][B][E]
//   K0b: pack 8 recurrent weight mats [256,1024] fp32 -> bf16 WMMA B-fragments
//   K1 : time-parallel GEMM  xW1 = xs @ W1 + b1  (both dirs), stored in
//        WMMA C-fragment layout (v8f per lane) for single-vector reload in K2
//   K2 : persistent recurrence, 1 WGP per direction, h in LDS, c in regs,
//        V_WMMA_F32_16X16X32_BF16; xW1 fragment loads software-pipelined
//        behind the h1@U1 matmul
//   K3 : dense(2H->5) + softmax, VALU + wave reduction
// ---------------------------------------------------------------------------

#define TT   512
#define BB   64
#define EE   256
#define HH   256
#define G4   1024          // 4*H
#define MM   (TT*BB)       // 32768 rows in the time-parallel GEMM
#define MT2  2048          // MM/16 mTiles
#define NT2  64            // G4/16 nTiles

typedef __bf16 bf16_t;
typedef __attribute__((ext_vector_type(8)))  __bf16 v8bf;
typedef __attribute__((ext_vector_type(16))) __bf16 v16bf;
typedef __attribute__((ext_vector_type(8)))  float  v8f;

// ---------------- WMMA helpers ----------------

__device__ __forceinline__ v8f wmma_bf16(v16bf a, v16bf b, v8f c) {
  // D = A(16x32 bf16) * B(32x16 bf16) + C(16x16 f32)
  return __builtin_amdgcn_wmma_f32_16x16x32_bf16(
      /*neg_a=*/false, a, /*neg_b=*/false, b,
      /*c_mod=*/(short)0, c, /*reuse_a=*/false, /*reuse_b=*/false);
}

// A fragment (16x32 bf16) from row-major [*, ld] bf16 buffer.
__device__ __forceinline__ v16bf load_a_frag(const bf16_t* base, int row0,
                                             int ld, int kBase, int lane) {
  const int m    = row0 + (lane & 15);
  const int half = lane >> 4;
  const bf16_t* p = base + (size_t)m * ld + kBase + half * 8;
  v8bf lo = *(const v8bf*)p;
  v8bf hi = *(const v8bf*)(p + 16);
  v16bf r;
#pragma unroll
  for (int i = 0; i < 8; ++i) { r[i] = lo[i]; r[i + 8] = hi[i]; }
  return r;
}

// B fragment (32x16 bf16) from the packed-weight buffer.
// Pack layout: frag = ((mat*512 + kT*64 + nT)*32 + lane)*16 bf16 elems.
__device__ __forceinline__ v16bf load_b_frag(const bf16_t* pack, int matId,
                                             int kT, int nT, int lane) {
  const size_t fi = (size_t)matId * 512 + (size_t)kT * 64 + (size_t)nT;
  const bf16_t* p = pack + (fi * 32 + (size_t)lane) * 16;
  return *(const v16bf*)p;
}

__device__ __forceinline__ float sigmoidf_(float x) {
  return 1.0f / (1.0f + __expf(-x));
}
__device__ __forceinline__ float tanhf_(float x) {
  // 1 - 2/(e^{2x}+1): one v_exp + one rcp, saturates correctly at +/-inf
  float e = __expf(2.0f * x);
  return 1.0f - 2.0f / (e + 1.0f);
}

// ---------------- K0a: embedding gather -> bf16 xs[T][B][E] ----------------

__global__ void k_embed_gather(const int* __restrict__ tokens,
                               const float* __restrict__ emb,
                               bf16_t* __restrict__ xs) {
  int idx = blockIdx.x * blockDim.x + threadIdx.x;     // < T*B*E = 8388608
  if (idx >= TT * BB * EE) return;
  int e   = idx & 255;
  int pos = idx >> 8;           // t*64 + b
  int t   = pos >> 6;
  int b   = pos & 63;
  int tok = tokens[b * TT + t];
  xs[idx] = (bf16_t)emb[(size_t)tok * EE + e];
}

// ---------------- K0b: pack weights fp32 -> bf16 B-fragments ----------------

struct WPtrs { const float* m[8]; };   // fwW1 fwU1 fwW2 fwU2 bwW1 bwU1 bwW2 bwU2

__global__ void k_pack_weights(WPtrs ptrs, bf16_t* __restrict__ wpack) {
  int idx = blockIdx.x * blockDim.x + threadIdx.x;     // < 8*262144
  if (idx >= 8 * 262144) return;
  int mat  = idx >> 18;
  int rem  = idx & 262143;
  int f    = rem >> 9;          // kT*64 + nT
  int le   = rem & 511;         // lane*16 + e
  int lane = le >> 4;
  int e    = le & 15;
  int kT   = f >> 6;
  int nT   = f & 63;
  int K    = kT * 32 + ((lane & 16) ? 8 : 0) + (e & 7) + ((e & 8) ? 16 : 0);
  int n    = nT * 16 + (lane & 15);
  wpack[idx] = (bf16_t)ptrs.m[mat][(size_t)K * G4 + n];
}

// ---------------- K1: xW1 = xs @ W1 + b1 (both directions) ----------------
// Output layout: fragment-major. frag f = mT*64 + nT; per-dir base d*MT2*NT2;
// element = (base + f)*256 + lane*8  (v8f per lane, C-matrix layout).

__global__ void k_xw1_gemm(const bf16_t* __restrict__ xs,
                           const bf16_t* __restrict__ wpack,
                           const float* __restrict__ b1f,
                           const float* __restrict__ b1b,
                           float* __restrict__ xW1) {
  const int wave = threadIdx.x >> 5;
  const int lane = threadIdx.x & 31;
  unsigned tile = blockIdx.x * 8u + wave;   // 2 dirs * 2048 mTiles * 64 nTiles
  const int d   = (int)(tile >> 17);
  const int rem = (int)(tile & 131071u);
  const int mT  = rem >> 6;
  const int nT  = rem & 63;

  const float* bias = d ? b1b : b1f;
  const int matId   = d ? 4 : 0;            // *_W1
  const int col     = nT * 16 + (lane & 15);

  v8f acc;
  {
    float bv = bias[col];
#pragma unroll
    for (int e = 0; e < 8; ++e) acc[e] = bv;
  }

#pragma unroll
  for (int kT = 0; kT < 8; ++kT) {
    v16bf a = load_a_frag(xs, mT * 16, EE, kT * 32, lane);
    v16bf b = load_b_frag(wpack, matId, kT, nT, lane);
    acc = wmma_bf16(a, b, acc);
  }

  float* fp = xW1 + ((size_t)d * MT2 * NT2 + (size_t)mT * NT2 + nT) * 256 +
              lane * 8;
  *(v8f*)fp = acc;                          // 2x global_store_b128, coalesced
}

// ---------------- K2: persistent recurrence ----------------

__device__ __forceinline__ void mm_step(v8f (&acc)[2][4],
                                        const bf16_t* __restrict__ hbuf,
                                        const bf16_t* __restrict__ wpack,
                                        int matId, int mp, int hT, int lane) {
#pragma unroll
  for (int kT = 0; kT < 8; ++kT) {
    v16bf a0 = load_a_frag(hbuf, (mp * 2 + 0) * 16, HH, kT * 32, lane);
    v16bf a1 = load_a_frag(hbuf, (mp * 2 + 1) * 16, HH, kT * 32, lane);
    if (kT < 7) {
      // warm L2->L0 path for next K-slice of the weight stream
      const bf16_t* nb =
          wpack + (((size_t)matId * 512 + (size_t)(kT + 1) * 64 + hT) * 32 +
                   (size_t)lane) * 16;
      __builtin_prefetch(nb, 0, 0);
    }
#pragma unroll
    for (int g = 0; g < 4; ++g) {
      v16bf b = load_b_frag(wpack, matId, kT, g * 16 + hT, lane);
      acc[0][g] = wmma_bf16(a0, b, acc[0][g]);
      acc[1][g] = wmma_bf16(a1, b, acc[1][g]);
    }
  }
}

__device__ __forceinline__ void cell_update(v8f (&z)[4], v8f& c,
                                            bf16_t* __restrict__ hbuf,
                                            float* __restrict__ yout,
                                            int mTile, int hT, int lane) {
  const int col     = hT * 16 + (lane & 15);
  const int rowBase = mTile * 16 + ((lane & 16) ? 8 : 0);
#pragma unroll
  for (int e = 0; e < 8; ++e) {
    float iv = sigmoidf_(z[0][e]);
    float fv = sigmoidf_(z[1][e]);
    float gv = tanhf_(z[2][e]);
    float ov = sigmoidf_(z[3][e]);
    float cn = fv * c[e] + iv * gv;
    c[e] = cn;
    float hv = ov * tanhf_(cn);
    int ofs = (rowBase + e) * HH + col;
    hbuf[ofs] = (bf16_t)hv;
    if (yout) yout[ofs] = hv;
  }
}

__global__ void __launch_bounds__(1024, 1)
k_recurrent(const float* __restrict__ xW1, const bf16_t* __restrict__ wpack,
            const float* __restrict__ b1f, const float* __restrict__ b2f,
            const float* __restrict__ b1b, const float* __restrict__ b2b,
            float* __restrict__ ys) {
  __shared__ bf16_t h1s[BB * HH];   // 32 KB
  __shared__ bf16_t h2s[BB * HH];   // 32 KB

  const int dir  = blockIdx.x;      // 0 = fw, 1 = bw
  const int tid  = threadIdx.x;
  const int wave = tid >> 5;
  const int lane = tid & 31;
  const int hT   = wave & 15;       // hidden-column tile (0..15)
  const int mp   = wave >> 4;       // batch pair (0..1) -> mTiles {2mp, 2mp+1}

  const float* b1  = dir ? b1b : b1f;
  const float* b2  = dir ? b2b : b2f;
  const int mU1 = dir ? 5 : 1, mW2 = dir ? 6 : 2, mU2 = dir ? 7 : 3;
  const float* xW  = xW1 + (size_t)dir * ((size_t)MT2 * NT2 * 256);
  float* ysd       = ys + (size_t)dir * ((size_t)TT * BB * HH);

  for (int i = tid; i < BB * HH; i += 1024) {
    h1s[i] = (bf16_t)0.f;
    h2s[i] = (bf16_t)0.f;
  }

  v8f c1[2], c2[2];
#pragma unroll
  for (int p = 0; p < 2; ++p)
#pragma unroll
    for (int e = 0; e < 8; ++e) { c1[p][e] = 0.f; c2[p][e] = 0.f; }

  __syncthreads();

  for (int step = 0; step < TT; ++step) {
    const int t = dir ? (TT - 1 - step) : step;

    // ---- layer 1: z1 = xW1[t] + b1 + h1 @ U1 ----
    // Issue the xW1 C-fragment loads FIRST; they complete behind the
    // h1@U1 WMMA stream and are folded in afterwards (latency hidden).
    v8f xfrag[2][4];
#pragma unroll
    for (int p = 0; p < 2; ++p)
#pragma unroll
      for (int g = 0; g < 4; ++g) {
        const size_t f =
            (size_t)(t * 4 + mp * 2 + p) * NT2 + (size_t)(g * 16 + hT);
        xfrag[p][g] = *(const v8f*)(xW + f * 256 + lane * 8);
      }

    v8f acc[2][4];
#pragma unroll
    for (int p = 0; p < 2; ++p)
#pragma unroll
      for (int g = 0; g < 4; ++g) {
        const float bv = b1[g * 256 + hT * 16 + (lane & 15)];
#pragma unroll
        for (int e = 0; e < 8; ++e) acc[p][g][e] = bv;
      }

    mm_step(acc, h1s, wpack, mU1, mp, hT, lane);

#pragma unroll
    for (int p = 0; p < 2; ++p)
#pragma unroll
      for (int g = 0; g < 4; ++g)
#pragma unroll
        for (int e = 0; e < 8; ++e) acc[p][g][e] += xfrag[p][g][e];

    __syncthreads();                      // all reads of h1 done
#pragma unroll
    for (int p = 0; p < 2; ++p)
      cell_update(acc[p], c1[p], h1s, nullptr, mp * 2 + p, hT, lane);
    __syncthreads();                      // new h1 visible

    // ---- layer 2: z2 = b2 + h1 @ W2 + h2 @ U2 ----
#pragma unroll
    for (int p = 0; p < 2; ++p)
#pragma unroll
      for (int g = 0; g < 4; ++g) {
        const float bv = b2[g * 256 + hT * 16 + (lane & 15)];
#pragma unroll
        for (int e = 0; e < 8; ++e) acc[p][g][e] = bv;
      }
    mm_step(acc, h1s, wpack, mW2, mp, hT, lane);
    mm_step(acc, h2s, wpack, mU2, mp, hT, lane);
    __syncthreads();                      // all reads of h2 done
    float* yo = ysd + (size_t)t * (BB * HH);
#pragma unroll
    for (int p = 0; p < 2; ++p)
      cell_update(acc[p], c2[p], h2s, yo, mp * 2 + p, hT, lane);
    __syncthreads();                      // new h2 visible
  }
}

// ---------------- K3: dense(512->5) + softmax ----------------

__global__ void k_dense_softmax(const float* __restrict__ ys,
                                const float* __restrict__ dW,
                                const float* __restrict__ db,
                                float* __restrict__ out) {
  const int wave = threadIdx.x >> 5;
  const int lane = threadIdx.x & 31;
  const int pos  = blockIdx.x * 8 + wave;   // b*T + t, < 32768
  const int b = pos >> 9;
  const int t = pos & 511;

  const float* fw = ys + ((size_t)t * BB + b) * HH;
  const float* bw = ys + (size_t)TT * BB * HH + ((size_t)t * BB + b) * HH;

  float acc[5] = {0.f, 0.f, 0.f, 0.f, 0.f};
#pragma unroll
  for (int it = 0; it < 8; ++it) {
    const int k = lane + it * 32;
    const float a  = fw[k];
    const float bb = bw[k];
#pragma unroll
    for (int j = 0; j < 5; ++j)
      acc[j] += a * dW[k * 5 + j] + bb * dW[(k + 256) * 5 + j];
  }
#pragma unroll
  for (int off = 16; off > 0; off >>= 1)
#pragma unroll
    for (int j = 0; j < 5; ++j) acc[j] += __shfl_down(acc[j], off, 32);

  if (lane == 0) {
    float l[5], m = -1e30f;
#pragma unroll
    for (int j = 0; j < 5; ++j) { l[j] = acc[j] + db[j]; m = fmaxf(m, l[j]); }
    float s = 0.f;
#pragma unroll
    for (int j = 0; j < 5; ++j) { l[j] = __expf(l[j] - m); s += l[j]; }
    const float inv = 1.f / s;
    float* o = out + ((size_t)b * TT + t) * 5;
#pragma unroll
    for (int j = 0; j < 5; ++j) o[j] = l[j] * inv;
  }
}

// ---------------- host launcher ----------------

extern "C" void kernel_launch(void* const* d_in, const int* in_sizes, int n_in,
                              void* d_out, int out_size, void* d_ws,
                              size_t ws_size, hipStream_t stream) {
  const int*   tokens = (const int*)d_in[0];
  const float* emb    = (const float*)d_in[1];
  const float* fw_W1  = (const float*)d_in[2];
  const float* fw_U1  = (const float*)d_in[3];
  const float* fw_b1  = (const float*)d_in[4];
  const float* fw_W2  = (const float*)d_in[5];
  const float* fw_U2  = (const float*)d_in[6];
  const float* fw_b2  = (const float*)d_in[7];
  const float* bw_W1  = (const float*)d_in[8];
  const float* bw_U1  = (const float*)d_in[9];
  const float* bw_b1  = (const float*)d_in[10];
  const float* bw_W2  = (const float*)d_in[11];
  const float* bw_U2  = (const float*)d_in[12];
  const float* bw_b2  = (const float*)d_in[13];
  const float* denseW = (const float*)d_in[14];
  const float* denseB = (const float*)d_in[15];
  float* out = (float*)d_out;

  char* wsb = (char*)d_ws;
  size_t o = 0;
  bf16_t* xs    = (bf16_t*)(wsb + o); o += (size_t)TT * BB * EE * sizeof(bf16_t);     // 16 MB
  bf16_t* wpack = (bf16_t*)(wsb + o); o += (size_t)8 * 262144 * sizeof(bf16_t);       // 4 MB
  o = (o + 255) & ~(size_t)255;
  float* xW1 = (float*)(wsb + o); o += (size_t)2 * MM * G4 * sizeof(float);           // 256 MB
  float* ys  = (float*)(wsb + o); o += (size_t)2 * TT * BB * HH * sizeof(float);      // 64 MB
  (void)ws_size; (void)n_in; (void)in_sizes; (void)out_size;

  k_embed_gather<<<(TT * BB * EE) / 256, 256, 0, stream>>>(tokens, emb, xs);

  WPtrs wp;
  wp.m[0] = fw_W1; wp.m[1] = fw_U1; wp.m[2] = fw_W2; wp.m[3] = fw_U2;
  wp.m[4] = bw_W1; wp.m[5] = bw_U1; wp.m[6] = bw_W2; wp.m[7] = bw_U2;
  k_pack_weights<<<(8 * 262144) / 256, 256, 0, stream>>>(wp, wpack);

  // 2 dirs * (32768/16) mTiles * (1024/16) nTiles = 262144 tiles, 8 waves/block
  k_xw1_gemm<<<262144 / 8, 256, 0, stream>>>(xs, wpack, fw_b1, bw_b1, xW1);

  k_recurrent<<<2, 1024, 0, stream>>>(xW1, wpack, fw_b1, fw_b2, bw_b1, bw_b2, ys);

  k_dense_softmax<<<(BB * TT) / 8, 256, 0, stream>>>(ys, denseW, denseB, out);
}
